// PointnetMeanShift_4209067950527
// MI455X (gfx1250) — matrix operand order repaired and checked by reading, not compiled, not run on hardware
//
#include <hip/hip_runtime.h>

typedef __attribute__((ext_vector_type(16))) _Float16 v16h;
typedef __attribute__((ext_vector_type(8)))  _Float16 v8h;
typedef __attribute__((ext_vector_type(2)))  _Float16 v2h;
typedef __attribute__((ext_vector_type(8)))  float    v8f;

#define NB     4
#define NP     2048
#define CF     128
#define NSAMP  64
#define R2     0.0625f
#define K1PAD  160     // 131 padded to multiple of 32
#define LDH    72      // LDS row stride in halves (144B, 16B-aligned)

// ---------------- kernel 1: fold BN scale into weights, convert to f16 ----------------
__global__ void msprep_kernel(const float* __restrict__ W1, const float* __restrict__ W2,
                              const float* __restrict__ g1, const float* __restrict__ g2,
                              _Float16* __restrict__ W1h, _Float16* __restrict__ W2h) {
  int t = blockIdx.x * blockDim.x + threadIdx.x;
  float inv = rsqrtf(1.0f + 1e-5f);
  if (t < 256 * K1PAD) {
    int o = t / K1PAD, k = t - o * K1PAD;
    float v = (k < 131) ? W1[o * 131 + k] * g1[o] * inv : 0.0f;
    W1h[t] = (_Float16)v;
  }
  if (t < 128 * 256) {
    int o = t >> 8, k = t & 255;
    W2h[t] = (_Float16)(W2[o * 256 + k] * g2[o] * inv);
  }
}

// ---------------- kernel 2: ball query (wave32 per query point) ----------------
__global__ __launch_bounds__(256) void msbq_kernel(const float* __restrict__ xyz,
                                                   int* __restrict__ idxbuf) {
  int wid  = (blockIdx.x * blockDim.x + threadIdx.x) >> 5;   // one wave per point
  int lane = threadIdx.x & 31;
  if (wid >= NB * NP) return;
  int b = wid >> 11, n = wid & (NP - 1);
  const float* xb = xyz + (size_t)b * NP * 3;
  float qx = xb[n * 3 + 0], qy = xb[n * 3 + 1], qz = xb[n * 3 + 2];
  float qn = qx * qx + qy * qy + qz * qz;
  int base = wid * NSAMP;
  int cnt = 0, first = n;
  bool gotfirst = false;
  for (int mb = 0; mb < NP; mb += 32) {
    int m = mb + lane;
    float px = xb[m * 3 + 0], py = xb[m * 3 + 1], pz = xb[m * 3 + 2];
    float pm = px * px + py * py + pz * pz;
    float d2 = qn + pm - 2.0f * (qx * px + qy * py + qz * pz);  // reference formula
    bool pred = d2 < R2;
    unsigned mask = (unsigned)__ballot(pred);
    if (!gotfirst && mask) { first = mb + (__ffs((int)mask) - 1); gotfirst = true; }
    int rank = __popc(mask & ((1u << lane) - 1u));
    if (pred && (cnt + rank) < NSAMP) idxbuf[base + cnt + rank] = m;
    cnt += __popc(mask);
    if (cnt >= NSAMP) break;
  }
  for (int s = cnt + lane; s < NSAMP; s += 32) idxbuf[base + s] = first;
}

// ---------------- WMMA fragment loaders (CDNA5 16-bit layouts) ----------------
// A (16x32 MxK): lanes 0-15 row M=lane, elems 0..7 = K kbase+0..7, elems 8..15 = K kbase+16..23;
//                lanes 16-31 same rows, K offset +8.
static __device__ __forceinline__ v16h load_a16(const _Float16* __restrict__ W, int ldk,
                                                int mbase, int kbase, int lane) {
  int row  = mbase + (lane & 15);
  int koff = kbase + ((lane >> 4) << 3);
  const _Float16* p = W + row * ldk + koff;
  v8h lo = *(const v8h*)(p);
  v8h hi = *(const v8h*)(p + 16);
  return __builtin_shufflevector(lo, hi, 0,1,2,3,4,5,6,7,8,9,10,11,12,13,14,15);
}
// B (32x16 KxN): lane%16 = N column nbase+lane%16; elems 0..7 = K kbase+8h+0..7, elems 8..15 = K kbase+16+8h+0..7 (h = lane/16).
static __device__ __forceinline__ v16h load_b16(const _Float16* S, int kbase, int nbase, int lane) {
  const _Float16* p = S + (kbase + ((lane >> 4) << 3)) * LDH + nbase + (lane & 15);
  v16h r;
#pragma unroll
  for (int s = 0; s < 8; ++s) { r[s] = p[s * LDH]; r[s + 8] = p[(16 + s) * LDH]; }
  return r;
}

// ---------------- kernel 3: fused gather + MLP (WMMA) + reduction ----------------
__global__ __launch_bounds__(256) void msmain_kernel(
    const float* __restrict__ xyz, const float* __restrict__ feat,
    const float* __restrict__ W3, const float* __restrict__ g3,
    const float* __restrict__ b1, const float* __restrict__ b2, const float* __restrict__ b3,
    const int* __restrict__ idxbuf, const _Float16* __restrict__ W1h,
    const _Float16* __restrict__ W2h, float* __restrict__ out) {
  // LDS (61 KB): Hs(160x72 h)=23040 | Y1s(256x72 h)=36864 | Gs(3x64 f)=768 | y3s(64 f)=256
  //              | cf(128 f)=512 | ps(4x64 f)=1024.  Y2s (128x72 h) overlays dead Hs.
  __shared__ __align__(16) unsigned char smem[23040 + 36864 + 768 + 256 + 512 + 1024];
  _Float16* Hs  = (_Float16*)smem;
  _Float16* Y2s = (_Float16*)smem;
  _Float16* Y1s = (_Float16*)(smem + 23040);
  float*    Gs  = (float*)(smem + 59904);
  float*    y3s = (float*)(smem + 60672);
  float*    cf  = (float*)(smem + 60928);
  float*    ps  = (float*)(smem + 61440);

  int tid  = threadIdx.x;
  int lane = tid & 31;
  int w    = tid >> 5;
  int pid  = blockIdx.x;
  int b    = pid >> 11;
  int n    = pid & (NP - 1);

  // zero H with b128 stores (covers K-pad rows 131..159) + stage center features
  {
    v8h z = {};
    for (int i = tid; i < (K1PAD * LDH) / 8; i += 256) ((v8h*)Hs)[i] = z;
  }
  if (tid < CF) cf[tid] = feat[((size_t)b * CF + tid) * NP + n];
  __syncthreads();

  float cx = xyz[((size_t)b * NP + n) * 3 + 0];
  float cy = xyz[((size_t)b * NP + n) * 3 + 1];
  float cz = xyz[((size_t)b * NP + n) * 3 + 2];

  // gather: 8 threads per column PAIR; every H store is a packed 2xf16 b32 store
  {
    int cp  = tid >> 3;             // 0..31 column pair
    int sub = tid & 7;              // 0..7 -> 16 channels each
    int col0 = cp * 2, col1 = col0 + 1;
    int j0 = idxbuf[(size_t)pid * NSAMP + col0];
    int j1 = idxbuf[(size_t)pid * NSAMP + col1];
    float p0x = xyz[((size_t)b * NP + j0) * 3 + 0];
    float p0y = xyz[((size_t)b * NP + j0) * 3 + 1];
    float p0z = xyz[((size_t)b * NP + j0) * 3 + 2];
    float p1x = xyz[((size_t)b * NP + j1) * 3 + 0];
    float p1y = xyz[((size_t)b * NP + j1) * 3 + 1];
    float p1z = xyz[((size_t)b * NP + j1) * 3 + 2];
    if (sub == 0) {
      Gs[0 * NSAMP + col0] = p0x - cx;  Gs[0 * NSAMP + col1] = p1x - cx;
      Gs[1 * NSAMP + col0] = p0y - cy;  Gs[1 * NSAMP + col1] = p1y - cy;
      Gs[2 * NSAMP + col0] = p0z - cz;  Gs[2 * NSAMP + col1] = p1z - cz;
      float d0x = p0x - 2.0f * cx, d1x = p1x - 2.0f * cx;
      float d0y = p0y - 2.0f * cy, d1y = p1y - 2.0f * cy;
      float d0z = p0z - 2.0f * cz, d1z = p1z - 2.0f * cz;
      *(v2h*)&Hs[0 * LDH + col0] = (v2h){(_Float16)(d0x * d0x), (_Float16)(d1x * d1x)};
      *(v2h*)&Hs[1 * LDH + col0] = (v2h){(_Float16)(d0y * d0y), (_Float16)(d1y * d1y)};
      *(v2h*)&Hs[2 * LDH + col0] = (v2h){(_Float16)(d0z * d0z), (_Float16)(d1z * d1z)};
    }
#pragma unroll 4
    for (int k = 0; k < 16; ++k) {
      int ch = sub * 16 + k;
      float c  = cf[ch];
      float d0 = feat[((size_t)b * CF + ch) * NP + j0] - c;
      float d1 = feat[((size_t)b * CF + ch) * NP + j1] - c;
      *(v2h*)&Hs[(3 + ch) * LDH + col0] = (v2h){(_Float16)(d0 * d0), (_Float16)(d1 * d1)};
    }
  }
  __syncthreads();

  // ---- layer 1: Y1 = relu(W1h(256x160) @ H(160x64) + b1) ----
  // wave w owns M-tiles {2w, 2w+1}; B fragments shared across both M-tiles.
  {
    int mbase0 = (2 * w) * 16, mbase1 = mbase0 + 16;
    int hi8    = (lane >> 4) << 3;
    int mrow0  = mbase0 + hi8, mrow1 = mbase1 + hi8;
    v8f acc[2][4];
#pragma unroll
    for (int r = 0; r < 8; ++r) {
      float bv0 = b1[mrow0 + r], bv1 = b1[mrow1 + r];
#pragma unroll
      for (int nt = 0; nt < 4; ++nt) { acc[0][nt][r] = bv0; acc[1][nt][r] = bv1; }
    }
#pragma unroll
    for (int kb = 0; kb < 5; ++kb) {
      v16h a0 = load_a16(W1h, K1PAD, mbase0, kb * 32, lane);
      v16h a1 = load_a16(W1h, K1PAD, mbase1, kb * 32, lane);
#pragma unroll
      for (int nt = 0; nt < 4; ++nt) {
        v16h bf = load_b16(Hs, kb * 32, nt * 16, lane);
        acc[0][nt] = __builtin_amdgcn_wmma_f32_16x16x32_f16(false, a0, false, bf,
                                                            (short)0, acc[0][nt], false, false);
        acc[1][nt] = __builtin_amdgcn_wmma_f32_16x16x32_f16(false, a1, false, bf,
                                                            (short)0, acc[1][nt], false, false);
      }
    }
    int ncol0 = lane & 15;
#pragma unroll
    for (int nt = 0; nt < 4; ++nt)
#pragma unroll
      for (int r = 0; r < 8; ++r) {
        Y1s[(mrow0 + r) * LDH + nt * 16 + ncol0] = (_Float16)fmaxf(acc[0][nt][r], 0.0f);
        Y1s[(mrow1 + r) * LDH + nt * 16 + ncol0] = (_Float16)fmaxf(acc[1][nt][r], 0.0f);
      }
  }
  __syncthreads();

  // ---- layer 2: Y2 = relu(W2h(128x256) @ Y1(256x64) + b2), wave w owns M-tile w ----
  {
    int mbase = w * 16;
    int mrow  = mbase + ((lane >> 4) << 3);
    float bias[8];
#pragma unroll
    for (int r = 0; r < 8; ++r) bias[r] = b2[mrow + r];
    v8f acc[4];
#pragma unroll
    for (int nt = 0; nt < 4; ++nt)
#pragma unroll
      for (int r = 0; r < 8; ++r) acc[nt][r] = bias[r];
#pragma unroll
    for (int kb = 0; kb < 8; ++kb) {
      v16h a = load_a16(W2h, 256, mbase, kb * 32, lane);
#pragma unroll
      for (int nt = 0; nt < 4; ++nt) {
        v16h bf = load_b16(Y1s, kb * 32, nt * 16, lane);
        acc[nt] = __builtin_amdgcn_wmma_f32_16x16x32_f16(false, a, false, bf,
                                                         (short)0, acc[nt], false, false);
      }
    }
    int ncol0 = lane & 15;
#pragma unroll
    for (int nt = 0; nt < 4; ++nt)
#pragma unroll
      for (int r = 0; r < 8; ++r)
        Y2s[(mrow + r) * LDH + nt * 16 + ncol0] = (_Float16)fmaxf(acc[nt][r], 0.0f);
  }
  __syncthreads();

  // ---- layer 3: all 256 threads; 4-way split over channels + LDS partial reduction ----
  {
    int s = tid & 63, q = tid >> 6;
    float acc = 0.0f;
#pragma unroll 8
    for (int k = 0; k < 32; ++k) {
      int c = q * 32 + k;
      acc += W3[c] * (float)Y2s[c * LDH + s];
    }
    ps[q * 64 + s] = acc;
  }
  __syncthreads();
  if (tid < NSAMP) {
    float a = ps[tid] + ps[64 + tid] + ps[128 + tid] + ps[192 + tid];
    float inv = rsqrtf(1.0f + 1e-5f);
    y3s[tid] = fmaxf(g3[0] * inv * a + b3[0], 0.0f);
  }
  __syncthreads();
  if (tid < 3) {
    float num = 0.0f, den = 0.0f;
    for (int s = 0; s < NSAMP; ++s) { num += Gs[tid * NSAMP + s] * y3s[s]; den += y3s[s]; }
    out[((size_t)b * 3 + tid) * NP + n] = num / den;
  }
}

// ---------------- host launch ----------------
extern "C" void kernel_launch(void* const* d_in, const int* in_sizes, int n_in,
                              void* d_out, int out_size, void* d_ws, size_t ws_size,
                              hipStream_t stream) {
  const float* xyz  = (const float*)d_in[0];
  const float* feat = (const float*)d_in[1];
  const float* W1   = (const float*)d_in[2];
  const float* W2   = (const float*)d_in[3];
  const float* W3   = (const float*)d_in[4];
  const float* g1   = (const float*)d_in[5];
  const float* g2   = (const float*)d_in[6];
  const float* g3   = (const float*)d_in[7];
  const float* b1   = (const float*)d_in[8];
  const float* b2   = (const float*)d_in[9];
  const float* b3   = (const float*)d_in[10];
  float* out = (float*)d_out;

  unsigned char* ws = (unsigned char*)d_ws;
  int*      idxbuf = (int*)ws;                               // 4*2048*64*4 = 2 MiB
  _Float16* W1h    = (_Float16*)(ws + (size_t)NB * NP * NSAMP * 4);
  _Float16* W2h    = (_Float16*)(ws + (size_t)NB * NP * NSAMP * 4 + 256 * K1PAD * 2);

  msprep_kernel<<<(256 * K1PAD + 255) / 256, 256, 0, stream>>>(W1, W2, g1, g2, W1h, W2h);
  msbq_kernel<<<(NB * NP * 32) / 256, 256, 0, stream>>>(xyz, idxbuf);
  msmain_kernel<<<NB * NP, 256, 0, stream>>>(xyz, feat, W3, g3, b1, b2, b3,
                                             idxbuf, W1h, W2h, out);
}
